// TriGAT_1855425872580
// MI455X (gfx1250) — matrix operand order repaired and checked by reading, not compiled, not run on hardware
//
#include <hip/hip_runtime.h>
#include <cstdint>

typedef __attribute__((ext_vector_type(16))) _Float16 v16h;
typedef __attribute__((ext_vector_type(8)))  float    v8f;

// ---------- helpers ----------
__device__ __forceinline__ unsigned f2ord(float f) {
    unsigned u = __float_as_uint(f);
    return (u & 0x80000000u) ? ~u : (u | 0x80000000u);
}
__device__ __forceinline__ float ord2f(unsigned u) {
    return (u & 0x80000000u) ? __uint_as_float(u & 0x7fffffffu) : __uint_as_float(~u);
}
__device__ __forceinline__ float lrelu(float v) { return v > 0.f ? v : 0.2f * v; }

// ---------- generic utility kernels ----------
__global__ void k_fill_u32(unsigned* p, unsigned v, long n) {
    long t = (long)blockIdx.x * blockDim.x + threadIdx.x;
    if (t < n) p[t] = v;
}

__global__ void k_cvt_f32_f16(const float* __restrict__ in, _Float16* __restrict__ out, long n) {
    long t = (long)blockIdx.x * blockDim.x + threadIdx.x;
    if (t < n) out[t] = (_Float16)in[t];
}

// Wt[n*K + k] = (f16) W[k*Nw + n]   (W is K x Nw row-major)
__global__ void k_transpose_f16(const float* __restrict__ W, _Float16* __restrict__ Wt,
                                int K, int Nw) {
    long t = (long)blockIdx.x * blockDim.x + threadIdx.x;
    if (t >= (long)K * Nw) return;
    int n = (int)(t / K);
    int k = (int)(t % K);
    Wt[(size_t)n * K + k] = (_Float16)W[(size_t)k * Nw + n];
}

// ---------- WMMA GEMM: C[M x Nw] = A[M x K] * B[K x Nw], B given transposed (Bt[Nw x K]) ----------
// One wave per 16x16 output tile. K multiple of 32, Nw multiple of 16.
__global__ void k_gemm_wmma(const _Float16* __restrict__ A, const _Float16* __restrict__ Bt,
                            float* __restrict__ C, int M, int K, int Nw) {
    int wave = threadIdx.x >> 5;
    int lane = threadIdx.x & 31;
    int nTiles = Nw >> 4;
    int mTiles = (M + 15) >> 4;
    int tile = blockIdx.x * (blockDim.x >> 5) + wave;
    if (tile >= mTiles * nTiles) return;       // wave-uniform: EXEC stays all-ones for WMMA
    int mTile = tile / nTiles;
    int nTile = tile % nTiles;
    int l16   = lane & 15;
    bool hi   = lane >= 16;

    int row = mTile * 16 + l16; if (row >= M) row = M - 1;
    int bcol = nTile * 16 + l16;
    const _Float16* arow = A  + (size_t)row  * K;
    const _Float16* brow = Bt + (size_t)bcol * K;

    v8f acc = {};
    for (int k = 0; k < K; k += 32) {
        union { v16h v; float4 f[2]; } a, b;
        // A 16x32 f16 layout: lanes 0-15 hold K k..k+7 / k+16..k+23 ; lanes 16-31 hold K k+8..k+15 / k+24..k+31
        int ka = k + (hi ? 8 : 0);
        a.f[0] = *(const float4*)(arow + ka);
        a.f[1] = *(const float4*)(arow + ka + 16);
        // B 32x16 f16 layout: lanes 0-15 hold K k..k+7 / k+16..k+23 ; lanes 16-31 hold K k+8..k+15 / k+24..k+31 (N = lane&15)
        int kb = k + (hi ? 8 : 0);
        b.f[0] = *(const float4*)(brow + kb);
        b.f[1] = *(const float4*)(brow + kb + 16);
        acc = __builtin_amdgcn_wmma_f32_16x16x32_f16(false, a.v, false, b.v,
                                                     (short)0, acc, false, false);
    }
    // D layout: VGPR r -> (M = r or r+8, N = lane&15)
    int n = nTile * 16 + l16;
    int mBase = mTile * 16 + (hi ? 8 : 0);
    for (int r = 0; r < 8; ++r) {
        int m = mBase + r;
        if (m < M) C[(size_t)m * Nw + n] = acc[r];
    }
}

// ---------- attention coefficients: out[node*heads+h] = dot(xp[node, h*chid : (h+1)*chid], av[h]) ----------
__global__ void k_attn_coef(const float* __restrict__ xp, const float* __restrict__ av,
                            float* __restrict__ out, int n, int heads, int chid) {
    int t = blockIdx.x * blockDim.x + threadIdx.x;
    if (t >= n * heads) return;
    int node = t / heads, h = t % heads;
    const float* row = xp + (size_t)node * heads * chid + (size_t)h * chid;
    const float* a   = av + (size_t)h * chid;
    float acc = 0.f;
    for (int c = 0; c < chid; ++c) acc += row[c] * a[c];
    out[t] = acc;
}

// ---------- phase 1: segment max over dst (ordered-uint atomicMax) ----------
__global__ void k_edge_max(const int* __restrict__ src, const int* __restrict__ dst,
                           int E, int Nn, const float* __restrict__ as_,
                           const float* __restrict__ ad_, unsigned* __restrict__ mu, int heads) {
    int t = blockIdx.x * blockDim.x + threadIdx.x;
    if (t >= E + Nn) return;
    int si, di;
    if (t < E) { si = src[t]; di = dst[t]; } else { si = di = t - E; }
    for (int h = 0; h < heads; ++h) {
        float v = lrelu(as_[si * heads + h] + ad_[di * heads + h]);
        atomicMax(mu + (size_t)di * heads + h, f2ord(v));
    }
}

// ---------- phase 2: segment sum of exp(e - m) ----------
__global__ void k_edge_sum(const int* __restrict__ src, const int* __restrict__ dst,
                           int E, int Nn, const float* __restrict__ as_,
                           const float* __restrict__ ad_, const unsigned* __restrict__ mu,
                           float* __restrict__ s, int heads) {
    int t = blockIdx.x * blockDim.x + threadIdx.x;
    if (t >= E + Nn) return;
    int si, di;
    if (t < E) { si = src[t]; di = dst[t]; } else { si = di = t - E; }
    for (int h = 0; h < heads; ++h) {
        float v = lrelu(as_[si * heads + h] + ad_[di * heads + h]);
        float m = ord2f(mu[(size_t)di * heads + h]);
        atomicAdd(s + (size_t)di * heads + h, __expf(v - m));
    }
}

// ---------- phase 3 (convs): wave per edge, 128 features, heads=2 (head = lane>>4) ----------
__global__ void k_scatter_conv(const int* __restrict__ src, const int* __restrict__ dst,
                               int E, int Nn, const float* __restrict__ as_,
                               const float* __restrict__ ad_, const unsigned* __restrict__ mu,
                               const float* __restrict__ s, const float* __restrict__ xp,
                               float* __restrict__ hout, int colOff) {
    long gid = (long)blockIdx.x * blockDim.x + threadIdx.x;
    long e = gid >> 5;
    int lane = (int)(gid & 31);
    if (e >= (long)E + Nn) return;
    int si, di;
    if (e < E) { si = src[e]; di = dst[e]; } else { si = di = (int)(e - E); }
    int h = lane >> 4;                      // 64 feats per head, 4 feats per lane
    float v = lrelu(as_[(size_t)si * 2 + h] + ad_[(size_t)di * 2 + h]);
    float m = ord2f(mu[(size_t)di * 2 + h]);
    float alpha = __expf(v - m) / s[(size_t)di * 2 + h];
    float4 xv = *(const float4*)(xp + (size_t)si * 128 + lane * 4);
    float* o = hout + (size_t)di * 384 + colOff + lane * 4;
    atomicAdd(o + 0, alpha * xv.x);
    atomicAdd(o + 1, alpha * xv.y);
    atomicAdd(o + 2, alpha * xv.z);
    atomicAdd(o + 3, alpha * xv.w);
}

// ---------- phase 3 (final): 4 threads per edge, 16 features, heads=1 ----------
__global__ void k_scatter_final(const int* __restrict__ src, const int* __restrict__ dst,
                                int E, int Nn, const float* __restrict__ as_,
                                const float* __restrict__ ad_, const unsigned* __restrict__ mu,
                                const float* __restrict__ s, const float* __restrict__ hp,
                                float* __restrict__ o16) {
    long gid = (long)blockIdx.x * blockDim.x + threadIdx.x;
    long e = gid >> 2;
    int q = (int)(gid & 3);
    if (e >= (long)E + Nn) return;
    int si, di;
    if (e < E) { si = src[e]; di = dst[e]; } else { si = di = (int)(e - E); }
    float v = lrelu(as_[si] + ad_[di]);
    float alpha = __expf(v - ord2f(mu[di])) / s[di];
    float4 xv = *(const float4*)(hp + (size_t)si * 16 + q * 4);
    float* o = o16 + (size_t)di * 16 + q * 4;
    atomicAdd(o + 0, alpha * xv.x);
    atomicAdd(o + 1, alpha * xv.y);
    atomicAdd(o + 2, alpha * xv.z);
    atomicAdd(o + 3, alpha * xv.w);
}

// ---------- ELU(h + bias) -> f16 concat buffer ----------
__global__ void k_elu_bias_cvt(const float* __restrict__ h32, const float* __restrict__ b0,
                               const float* __restrict__ b1, const float* __restrict__ b2,
                               _Float16* __restrict__ hh, int n) {
    long t = (long)blockIdx.x * blockDim.x + threadIdx.x;
    if (t >= (long)n * 384) return;
    int c = (int)(t % 384);
    int cv = c >> 7, cc = c & 127;
    const float* bp = (cv == 0) ? b0 : ((cv == 1) ? b1 : b2);
    float v = h32[t] + bp[cc];
    v = v > 0.f ? v : (expf(v) - 1.f);
    hh[t] = (_Float16)v;
}

// ---------- softmax over 16 classes ----------
__global__ void k_softmax16(const float* __restrict__ o16, const float* __restrict__ bl,
                            float* __restrict__ out, int n) {
    int t = blockIdx.x * blockDim.x + threadIdx.x;
    if (t >= n) return;
    const float* r = o16 + (size_t)t * 16;
    float v[16], mx = -3.4e38f;
    for (int c = 0; c < 16; ++c) { v[c] = r[c] + bl[c]; mx = fmaxf(mx, v[c]); }
    float sum = 0.f;
    for (int c = 0; c < 16; ++c) { v[c] = expf(v[c] - mx); sum += v[c]; }
    float inv = 1.f / sum;
    for (int c = 0; c < 16; ++c) out[(size_t)t * 16 + c] = v[c] * inv;
}

// =====================================================================
extern "C" void kernel_launch(void* const* d_in, const int* in_sizes, int n_in,
                              void* d_out, int out_size, void* d_ws, size_t ws_size,
                              hipStream_t stream) {
    const float* x    = (const float*)d_in[0];
    const int*   eAll = (const int*)d_in[1];
    const int*   eSt  = (const int*)d_in[2];
    const int*   eTs  = (const int*)d_in[3];
    const float* Wc_[3] = {(const float*)d_in[4], (const float*)d_in[8],  (const float*)d_in[12]};
    const float* As_[3] = {(const float*)d_in[5], (const float*)d_in[9],  (const float*)d_in[13]};
    const float* Ad_[3] = {(const float*)d_in[6], (const float*)d_in[10], (const float*)d_in[14]};
    const float* Bi_[3] = {(const float*)d_in[7], (const float*)d_in[11], (const float*)d_in[15]};
    const float* Wl  = (const float*)d_in[16];
    const float* asl = (const float*)d_in[17];
    const float* adl = (const float*)d_in[18];
    const float* bl  = (const float*)d_in[19];

    const int N    = in_sizes[0] / 128;   // 50000
    const int EALL = in_sizes[1] / 2;     // 800000
    const int EST  = in_sizes[2] / 2;     // 400000
    const int ETS  = in_sizes[3] / 2;     // 400000

    // workspace layout
    char* ws = (char*)d_ws;
    size_t off = 0;
    auto alloc = [&](size_t bytes) -> char* {
        char* p = ws + off;
        off = (off + bytes + 255) & ~(size_t)255;
        return p;
    };
    _Float16* xh  = (_Float16*)alloc((size_t)N * 128 * 2);   // x in f16
    _Float16* wt  = (_Float16*)alloc((size_t)128 * 128 * 2); // transposed weights (reused)
    float*    xp  = (float*)   alloc((size_t)N * 128 * 4);   // per-conv xp; reused as hp (N x 16)
    float*    asb = (float*)   alloc((size_t)N * 2 * 4);
    float*    adb = (float*)   alloc((size_t)N * 2 * 4);
    unsigned* mu  = (unsigned*)alloc((size_t)N * 2 * 4);
    float*    sb  = (float*)   alloc((size_t)N * 2 * 4);
    float*    h32 = (float*)   alloc((size_t)N * 384 * 4);   // scatter accumulator (concat)
    _Float16* hh  = (_Float16*)alloc((size_t)N * 384 * 2);   // ELU(h) in f16
    float*    o16 = (float*)   alloc((size_t)N * 16 * 4);    // final-layer accumulator

    dim3 B(256);
    auto G = [](long n) { return dim3((unsigned)((n + 255) / 256)); };

    const unsigned ORD_NEG_INF = 0x007FFFFFu;  // f2ord(-inf)

    // x -> f16 ; zero concat accumulator
    k_cvt_f32_f16<<<G((long)N * 128), B, 0, stream>>>(x, xh, (long)N * 128);
    k_fill_u32<<<G((long)N * 384), B, 0, stream>>>((unsigned*)h32, 0u, (long)N * 384);

    const int* srcs[3] = {eSt, eTs, eAll};
    const int* dsts[3] = {eSt + EST, eTs + ETS, eAll + EALL};
    int        Es[3]   = {EST, ETS, EALL};

    int mTiles = (N + 15) / 16;
    for (int cv = 0; cv < 3; ++cv) {
        k_transpose_f16<<<G(128 * 128), B, 0, stream>>>(Wc_[cv], wt, 128, 128);
        int tiles = mTiles * (128 / 16);
        k_gemm_wmma<<<dim3((tiles + 7) / 8), B, 0, stream>>>(xh, wt, xp, N, 128, 128);
        k_attn_coef<<<G((long)N * 2), B, 0, stream>>>(xp, As_[cv], asb, N, 2, 64);
        k_attn_coef<<<G((long)N * 2), B, 0, stream>>>(xp, Ad_[cv], adb, N, 2, 64);
        k_fill_u32<<<G((long)N * 2), B, 0, stream>>>(mu, ORD_NEG_INF, (long)N * 2);
        k_fill_u32<<<G((long)N * 2), B, 0, stream>>>((unsigned*)sb, 0u, (long)N * 2);
        long tot = (long)Es[cv] + N;
        k_edge_max<<<G(tot), B, 0, stream>>>(srcs[cv], dsts[cv], Es[cv], N, asb, adb, mu, 2);
        k_edge_sum<<<G(tot), B, 0, stream>>>(srcs[cv], dsts[cv], Es[cv], N, asb, adb, mu, sb, 2);
        k_scatter_conv<<<G(tot * 32), B, 0, stream>>>(srcs[cv], dsts[cv], Es[cv], N,
                                                      asb, adb, mu, sb, xp, h32, cv * 128);
    }

    // ELU + bias + convert concat -> f16
    k_elu_bias_cvt<<<G((long)N * 384), B, 0, stream>>>(h32, Bi_[0], Bi_[1], Bi_[2], hh, N);

    // final GAT layer: hp = hh @ W_l  (K=384, N=16)
    k_transpose_f16<<<G(384 * 16), B, 0, stream>>>(Wl, wt, 384, 16);
    k_gemm_wmma<<<dim3((mTiles + 7) / 8), B, 0, stream>>>(hh, wt, xp, N, 384, 16);
    k_attn_coef<<<G((long)N), B, 0, stream>>>(xp, asl, asb, N, 1, 16);
    k_attn_coef<<<G((long)N), B, 0, stream>>>(xp, adl, adb, N, 1, 16);
    k_fill_u32<<<G((long)N), B, 0, stream>>>(mu, ORD_NEG_INF, (long)N);
    k_fill_u32<<<G((long)N), B, 0, stream>>>((unsigned*)sb, 0u, (long)N);
    k_fill_u32<<<G((long)N * 16), B, 0, stream>>>((unsigned*)o16, 0u, (long)N * 16);
    long tot = (long)EALL + N;
    k_edge_max<<<G(tot), B, 0, stream>>>(eAll, eAll + EALL, EALL, N, asb, adb, mu, 1);
    k_edge_sum<<<G(tot), B, 0, stream>>>(eAll, eAll + EALL, EALL, N, asb, adb, mu, sb, 1);
    k_scatter_final<<<G(tot * 4), B, 0, stream>>>(eAll, eAll + EALL, EALL, N,
                                                  asb, adb, mu, sb, xp, o16);

    // softmax -> d_out
    k_softmax16<<<G((long)N), B, 0, stream>>>(o16, bl, (float*)d_out, N);
}
